// GATLayer_66924180406944
// MI455X (gfx1250) — compile-verified, matrix operand order checked by reading, not run on hardware
//
#include <hip/hip_runtime.h>
#include <hip/hip_bf16.h>
#include <stdint.h>

// ---------------------------------------------------------------------------
// GAT layer for MI455X (gfx1250, wave32, WMMA).
//   N = E = 400000, HID = 128, HEADS = 4, K = 512
// Pipeline:
//   k_wconv    : W fp32 -> bf16 (one-time, 128x512)
//   k_nodepre  : per-node (one wave/node): xh = bf16(x), s_src/s_dst dots, denom=0
//   k_escore   : per-edge: leaky+head-softmax, ex=exp(p), atomicAdd denom[row]
//   k_ealpha   : per-edge: alpha = ex / denom[row]
//   k_gatwmma  : per 16-edge tile: gather x[col] bf16 -> A frags (shared by all
//                4 heads), W bf16 in LDS via TDM (padded rows), 128 x
//                v_wmma_f32_16x16x32_bf16 per tile, B streamed from LDS with
//                asm ds_load_b128 (prevents loop-invariant hoist -> spill),
//                alpha applied on the f32 accumulator per head,
//                + bias + residual, store out.
// ---------------------------------------------------------------------------

#define HID 128
#define HEADS 4
#define KTOT 512
#define NEG_SLOPE 0.01f

typedef __bf16 bf16_t;
typedef __attribute__((ext_vector_type(16))) __bf16 v16bf;
typedef __attribute__((ext_vector_type(8)))  float  v8f;
typedef __attribute__((ext_vector_type(4)))  unsigned int u32x4;
typedef __attribute__((ext_vector_type(8)))  int    i32x8;
typedef __attribute__((ext_vector_type(4)))  int    i32x4;

struct bf16x4 { bf16_t a, b, c, d; };   // 8-byte store unit

#define WLDS_STRIDE 520                  // 512 + 8 bf16 pad -> 4-bank rotation/row
#define WLDS_BYTES  (128 * WLDS_STRIDE * 2)

__device__ __forceinline__ float lrelu(float s) { return s > 0.f ? s : s * NEG_SLOPE; }

// ---------------------------------------------------------------- W -> bf16
__global__ void k_wconv(const float* __restrict__ W, bf16_t* __restrict__ Wh, int n4) {
  int i = blockIdx.x * blockDim.x + threadIdx.x;
  if (i >= n4) return;
  float4 v = ((const float4*)W)[i];
  bf16x4 o = { (bf16_t)v.x, (bf16_t)v.y, (bf16_t)v.z, (bf16_t)v.w };
  ((bf16x4*)Wh)[i] = o;
}

// ------------------------------------------- per-node: bf16 copy + 8 dots
__global__ void k_nodepre(const float* __restrict__ x, const float* __restrict__ att,
                          bf16_t* __restrict__ xh, float4* __restrict__ s_src,
                          float4* __restrict__ s_dst, float4* __restrict__ denom, int N) {
  const int wid  = blockIdx.x * (blockDim.x >> 5) + (threadIdx.x >> 5);
  const int lane = threadIdx.x & 31;
  if (wid >= N) return;

  const float* xr = x + (size_t)wid * HID;
  float4 xv = *(const float4*)(xr + 4 * lane);

  bf16x4 hv = { (bf16_t)xv.x, (bf16_t)xv.y, (bf16_t)xv.z, (bf16_t)xv.w };
  *(bf16x4*)(xh + (size_t)wid * HID + 4 * lane) = hv;

  float ps[HEADS], pd[HEADS];
#pragma unroll
  for (int h = 0; h < HEADS; ++h) {
    const float* ah = att + h * (2 * HID);
    float4 aS = *(const float4*)(ah + 4 * lane);
    float4 aD = *(const float4*)(ah + HID + 4 * lane);
    ps[h] = xv.x * aS.x + xv.y * aS.y + xv.z * aS.z + xv.w * aS.w;
    pd[h] = xv.x * aD.x + xv.y * aD.y + xv.z * aD.z + xv.w * aD.w;
  }
#pragma unroll
  for (int off = 16; off > 0; off >>= 1) {
#pragma unroll
    for (int h = 0; h < HEADS; ++h) {
      ps[h] += __shfl_xor(ps[h], off, 32);
      pd[h] += __shfl_xor(pd[h], off, 32);
    }
  }
  if (lane == 0) {
    s_src[wid] = make_float4(ps[0], ps[1], ps[2], ps[3]);
    s_dst[wid] = make_float4(pd[0], pd[1], pd[2], pd[3]);
    denom[wid] = make_float4(0.f, 0.f, 0.f, 0.f);
  }
}

// --------------------------- per-edge: head softmax, ex=exp(p), denom accum
__global__ void k_escore(const int* __restrict__ rowA, const int* __restrict__ colA,
                         const float4* __restrict__ s_src, const float4* __restrict__ s_dst,
                         float4* __restrict__ ex, float* __restrict__ denom, int E) {
  int e = blockIdx.x * blockDim.x + threadIdx.x;
  if (e >= E) return;
  int r = rowA[e], c = colA[e];
  float4 a = s_src[r], b = s_dst[c];
  float s0 = lrelu(a.x + b.x), s1 = lrelu(a.y + b.y);
  float s2 = lrelu(a.z + b.z), s3 = lrelu(a.w + b.w);
  float mx = fmaxf(fmaxf(s0, s1), fmaxf(s2, s3));
  float e0 = __expf(s0 - mx), e1 = __expf(s1 - mx), e2 = __expf(s2 - mx), e3 = __expf(s3 - mx);
  float inv = 1.f / (e0 + e1 + e2 + e3);
  // scatter-softmax numerators; p in (0,1) so the segment-max is dropped safely
  float x0 = __expf(e0 * inv), x1 = __expf(e1 * inv), x2 = __expf(e2 * inv), x3 = __expf(e3 * inv);
  ex[e] = make_float4(x0, x1, x2, x3);
  atomicAdd(denom + 4 * r + 0, x0);
  atomicAdd(denom + 4 * r + 1, x1);
  atomicAdd(denom + 4 * r + 2, x2);
  atomicAdd(denom + 4 * r + 3, x3);
}

// ------------------------------------------------ per-edge: normalize alpha
__global__ void k_ealpha(const int* __restrict__ rowA, float4* __restrict__ ex,
                         const float4* __restrict__ denom, int E) {
  int e = blockIdx.x * blockDim.x + threadIdx.x;
  if (e >= E) return;
  int r = rowA[e];
  float4 v = ex[e], d = denom[r];
  ex[e] = make_float4(v.x / d.x, v.y / d.y, v.z / d.z, v.w / d.w);
}

// --------------------------------------------------------- WMMA main kernel
__global__ void k_gatwmma(const bf16_t* __restrict__ xh, const bf16_t* __restrict__ Wh,
                          const float* __restrict__ alpha, const int* __restrict__ colA,
                          const float* __restrict__ x, const float* __restrict__ bias,
                          float* __restrict__ out, int E, int nTiles) {
  extern __shared__ __align__(16) char smem[];
  bf16_t* Wlds = (bf16_t*)smem;   // [128][WLDS_STRIDE] bf16, row j = W[j][0..511]

#if __has_builtin(__builtin_amdgcn_tensor_load_to_lds)
  // ---- TDM: DMA W (bf16, 128 rows x 512) into LDS with 8-element row pad.
  //      6-arg form (clang-23 / therock-10.0 toolchain): groups 0/1 used,
  //      groups 2/3 + trailing group zero (2D tensor), cpol = 0.
  if ((threadIdx.x >> 5) == 0) {
    unsigned long long ga = (unsigned long long)(uintptr_t)Wh;
    u32x4 g0 = { 1u,                                  // count = 1
                 0u,                                  // lds_addr = 0 (dynamic LDS base)
                 (unsigned)(ga & 0xFFFFFFFFu),
                 (unsigned)((ga >> 32) & 0x01FFFFFFu) | (2u << 30) };  // type=2
    // group1: data_size=2B, pad_enable, pad_interval=256 dwords, pad_amount=4 dwords
    int d0 = (int)((1u << 16) | (1u << 20) | (7u << 22) | (3u << 25));
    i32x8 g1 = { d0,
                 (int)(512u << 16),                    // tensor_dim0 = 512 (bits 48..63)
                 (int)(128u << 16),                    // tensor_dim1 = 128 (bits 80..95)
                 (int)(512u << 16),                    // tile_dim0   = 512 (bits 112..127)
                 (int)(128u),                          // tile_dim1   = 128
                 (int)(512u),                          // tensor_dim0_stride = 512
                 0, 0 };
    i32x4 z4 = { 0, 0, 0, 0 };
    i32x8 z8 = { 0, 0, 0, 0, 0, 0, 0, 0 };
    __builtin_amdgcn_tensor_load_to_lds(g0, g1, z4, z4, z8, 0);
    __builtin_amdgcn_s_wait_tensorcnt(0);
  }
#else
  // ---- Fallback: cooperative fill of padded W tile.
  for (int c = threadIdx.x; c < 128 * 64; c += blockDim.x) {
    int j = c >> 6, kc = (c & 63) << 3;
    *(u32x4*)(Wlds + j * WLDS_STRIDE + kc) = *(const u32x4*)(Wh + j * KTOT + kc);
  }
#endif
  __syncthreads();

  const int lane = threadIdx.x & 31;
  const int m    = lane & 15;      // A row / output column slot
  const int hi   = lane >> 4;      // half-wave selector in K / M
  const int wcnt = gridDim.x * (blockDim.x >> 5);
  int wid = blockIdx.x * (blockDim.x >> 5) + (threadIdx.x >> 5);

  for (int t = wid; t < nTiles; t += wcnt) {
    const int e0 = t * 16;
    __builtin_prefetch(alpha + (size_t)(e0 + 16 * wcnt) * 4, 0, 0);

    int ce = e0 + m; if (ce >= E) ce = E - 1;
    const bf16_t* xrow = xh + (size_t)colA[ce] * HID;

    // A fragments: 4 d-chunks of 32, shared by all 4 heads.
    // layout: elem i -> K = 16*(i/8) + 8*hi + (i%8) within the chunk.
    v16bf A[4];
#pragma unroll
    for (int c = 0; c < 4; ++c) {
      *((u32x4*)&A[c] + 0) = *(const u32x4*)(xrow + c * 32 + 8 * hi);
      *((u32x4*)&A[c] + 1) = *(const u32x4*)(xrow + c * 32 + 16 + 8 * hi);
    }

    // alpha rows for the 8 output rows this lane accumulates (M = r + 8*hi)
    float4 av[8];
#pragma unroll
    for (int r = 0; r < 8; ++r) {
      int er = e0 + 8 * hi + r; if (er >= E) er = E - 1;
      av[r] = *(const float4*)(alpha + (size_t)er * 4);
    }

#pragma unroll 1   // keep live set small: no cross-nt hoisting / scratch spills
    for (int nt = 0; nt < 8; ++nt) {
      const int j = nt * 16 + m;                         // output feature
      const unsigned wbase = (unsigned)j * (WLDS_STRIDE * 2) + 16u * (unsigned)hi;
      v8f outAcc = {0.f, 0.f, 0.f, 0.f, 0.f, 0.f, 0.f, 0.f};
#pragma unroll
      for (int h = 0; h < HEADS; ++h) {
        // B fragments for all 4 k-chunks of this head, streamed from LDS.
        // asm volatile: cannot be hoisted out of the tile loop (the hoist+spill
        // to scratch is exactly what we're preventing). One dscnt wait per 8
        // loads; WMMAs below depend on the outputs so they stay after the wait.
        const unsigned ab = wbase + (unsigned)(h * HID * 2);
        u32x4 b0l, b0h, b1l, b1h, b2l, b2h, b3l, b3h;
        asm volatile(
            "ds_load_b128 %0, %8\n\t"
            "ds_load_b128 %1, %9\n\t"
            "ds_load_b128 %2, %10\n\t"
            "ds_load_b128 %3, %11\n\t"
            "ds_load_b128 %4, %12\n\t"
            "ds_load_b128 %5, %13\n\t"
            "ds_load_b128 %6, %14\n\t"
            "ds_load_b128 %7, %15\n\t"
            "s_wait_dscnt 0x0"
            : "=&v"(b0l), "=&v"(b0h), "=&v"(b1l), "=&v"(b1h),
              "=&v"(b2l), "=&v"(b2h), "=&v"(b3l), "=&v"(b3h)
            : "v"(ab),        "v"(ab + 32u),  "v"(ab + 64u),  "v"(ab + 96u),
              "v"(ab + 128u), "v"(ab + 160u), "v"(ab + 192u), "v"(ab + 224u));
        v16bf B0, B1, B2, B3;
        *((u32x4*)&B0 + 0) = b0l; *((u32x4*)&B0 + 1) = b0h;
        *((u32x4*)&B1 + 0) = b1l; *((u32x4*)&B1 + 1) = b1h;
        *((u32x4*)&B2 + 0) = b2l; *((u32x4*)&B2 + 1) = b2h;
        *((u32x4*)&B3 + 0) = b3l; *((u32x4*)&B3 + 1) = b3h;

        v8f cacc = {0.f, 0.f, 0.f, 0.f, 0.f, 0.f, 0.f, 0.f};
        cacc = __builtin_amdgcn_wmma_f32_16x16x32_bf16(false, A[0], false, B0, (short)0, cacc, false, false);
        cacc = __builtin_amdgcn_wmma_f32_16x16x32_bf16(false, A[1], false, B1, (short)0, cacc, false, false);
        cacc = __builtin_amdgcn_wmma_f32_16x16x32_bf16(false, A[2], false, B2, (short)0, cacc, false, false);
        cacc = __builtin_amdgcn_wmma_f32_16x16x32_bf16(false, A[3], false, B3, (short)0, cacc, false, false);

#pragma unroll
        for (int r = 0; r < 8; ++r) {
          float a = (h == 0) ? av[r].x : (h == 1) ? av[r].y : (h == 2) ? av[r].z : av[r].w;
          outAcc[r] += a * cacc[r];
        }
      }
      const float bj = bias[j];
#pragma unroll
      for (int r = 0; r < 8; ++r) {
        int er = e0 + 8 * hi + r;
        if (er < E) {
          size_t oi = (size_t)er * HID + j;
          out[oi] = outAcc[r] + bj + x[oi];
        }
      }
    }
  }
}

// ---------------------------------------------------------------------------
extern "C" void kernel_launch(void* const* d_in, const int* in_sizes, int n_in,
                              void* d_out, int out_size, void* d_ws, size_t ws_size,
                              hipStream_t stream) {
  const float* x   = (const float*)d_in[0];
  const int*   ei  = (const int*)d_in[1];     // edge_index flat [2][E]
  const float* att = (const float*)d_in[2];   // [1, 4, 256]
  const float* W   = (const float*)d_in[3];   // [128, 512]
  const float* b   = (const float*)d_in[4];   // [128]
  const int N = in_sizes[0] / HID;
  const int E = in_sizes[1] / 2;
  const int* rowA = ei;
  const int* colA = ei + E;

  char* ws = (char*)d_ws;
  size_t off = 0;
  auto carve = [&](size_t bytes) { char* p = ws + off; off = (off + bytes + 255) & ~(size_t)255; return p; };
  float*  s_src = (float*)carve((size_t)N * 4 * sizeof(float));
  float*  s_dst = (float*)carve((size_t)N * 4 * sizeof(float));
  float*  denom = (float*)carve((size_t)N * 4 * sizeof(float));
  float*  exA   = (float*)carve((size_t)E * 4 * sizeof(float));  // ex -> alpha in place
  bf16_t* xh    = (bf16_t*)carve((size_t)N * HID * sizeof(bf16_t));
  bf16_t* Wh    = (bf16_t*)carve((size_t)HID * KTOT * sizeof(bf16_t));
  (void)ws_size; (void)n_in; (void)out_size;

  // 1) W -> bf16
  {
    int n4 = HID * KTOT / 4;
    k_wconv<<<(n4 + 255) / 256, 256, 0, stream>>>(W, Wh, n4);
  }
  // 2) per-node preprocessing: one wave per node
  {
    int wavesPerBlk = 256 / 32;
    int blocks = (N + wavesPerBlk - 1) / wavesPerBlk;
    k_nodepre<<<blocks, 256, 0, stream>>>(x, att, xh, (float4*)s_src, (float4*)s_dst,
                                          (float4*)denom, N);
  }
  // 3) per-edge scores + denominator accumulation
  k_escore<<<(E + 255) / 256, 256, 0, stream>>>(rowA, colA, (const float4*)s_src,
                                                (const float4*)s_dst, (float4*)exA, denom, E);
  // 4) normalize
  k_ealpha<<<(E + 255) / 256, 256, 0, stream>>>(rowA, (float4*)exA, (const float4*)denom, E);
  // 5) WMMA aggregation + linear + residual
  {
    int nTiles = (E + 15) / 16;
    (void)hipFuncSetAttribute((const void*)k_gatwmma,
                              hipFuncAttributeMaxDynamicSharedMemorySize, WLDS_BYTES);
    k_gatwmma<<<512, 128, WLDS_BYTES, stream>>>(xh, Wh, exA, colA, x, b,
                                                (float*)d_out, E, nTiles);
  }
}
